// RNN_13675175870864
// MI455X (gfx1250) — matrix operand order repaired and checked by reading, not compiled, not run on hardware
//
#include <hip/hip_runtime.h>

typedef __attribute__((ext_vector_type(16))) _Float16 v16h;
typedef __attribute__((ext_vector_type(8)))  float    v8f;

#define SEQ_N   1000000
#define NTILES  (SEQ_N / 16)

__device__ __forceinline__ float fast_tanh(float x) {
#if __has_builtin(__builtin_amdgcn_tanhf)
    return __builtin_amdgcn_tanhf(x);          // v_tanh_f32 on gfx1250
#else
    float e = __expf(2.0f * x);                // v_exp_f32 path
    return 1.0f - 2.0f / (e + 1.0f);           // saturates correctly at +/-1
#endif
}

__global__ __launch_bounds__(256) void rnn_fused_wmma(
    const float* __restrict__ x,
    const float* __restrict__ W_ih0,   // 64 x 1
    const float* __restrict__ b_ih0,   // 64
    const float* __restrict__ b_hh0,   // 64
    const float* __restrict__ W_ih1,   // 64 x 64
    const float* __restrict__ b_ih1,   // 64
    const float* __restrict__ b_hh1,   // 64
    const float* __restrict__ fc_w,    // 2 x 64
    const float* __restrict__ fc_b,    // 2
    float* __restrict__ out,           // SEQ x 2
    float* __restrict__ h0_out,        // SEQ x 64
    float* __restrict__ h1_out)        // SEQ x 64
{
    const int lane = threadIdx.x & 31;
    const int half = lane >> 4;        // which 16-lane group
    const int m16  = lane & 15;

    // ---- per-lane constants kept in VGPRs --------------------------------
    // Layer-0 weight/bias replicated into the A-fragment k-pattern:
    //   slot i = f*16 + 2v + e  ->  k = 32f + 2v + (v>=4 ? 8 : 0) + 8*half + e
    float ck[32], dk[32];
    #pragma unroll
    for (int f = 0; f < 2; ++f)
      #pragma unroll
      for (int v = 0; v < 8; ++v)
        #pragma unroll
        for (int e = 0; e < 2; ++e) {
            int k = 32*f + 2*v + (v >= 4 ? 8 : 0) + 8*half + e;
            int i = f*16 + 2*v + e;
            ck[i] = W_ih0[k];
            dk[i] = b_ih0[k] + b_hh0[k];
        }

    // B fragments: B[kt][ :, nt] of W_ih1^T, i.e. B[k][j] = W_ih1[j*64+k].
    // B 32x16 f16 layout: lanes 0-15 -> n=lane, K=0..7,16..23 (2 per VGPR);
    //                     lanes 16-31 -> n=lane-16, K=8..15,24..31.
    v16h Bf[4][2];
    #pragma unroll
    for (int t = 0; t < 4; ++t)
      #pragma unroll
      for (int c = 0; c < 2; ++c) {
        const int j = 16*t + m16;
        #pragma unroll
        for (int v = 0; v < 8; ++v) {
            const int k = 32*c + 2*v + (v >= 4 ? 8 : 0) + 8*half;
            Bf[t][c][2*v]   = (_Float16)W_ih1[j*64 + k];
            Bf[t][c][2*v+1] = (_Float16)W_ih1[j*64 + k + 1];
        }
      }

    // Layer-1 bias and fc weights for this lane's accumulator columns.
    float bs1[4], fw0[4], fw1[4];
    #pragma unroll
    for (int t = 0; t < 4; ++t) {
        const int col = 16*t + m16;
        bs1[t] = b_ih1[col] + b_hh1[col];
        fw0[t] = fc_w[col];
        fw1[t] = fc_w[64 + col];
    }
    const float fb0 = fc_b[0], fb1 = fc_b[1];

    const int waveId   = (int)((blockIdx.x * blockDim.x + threadIdx.x) >> 5);
    const int numWaves = (int)((gridDim.x * blockDim.x) >> 5);

    for (int tile = waveId; tile < NTILES; tile += numWaves) {
        const int rowbase = tile * 16;
        const int r  = rowbase + m16;           // lanes L and L+16 share row r
        const float xv = x[r];

        // ---- layer 0: h0 = tanh(x*w0 + b), built straight into A frags ---
        float h0v[32];
        #pragma unroll
        for (int i = 0; i < 32; ++i)
            h0v[i] = fast_tanh(fmaf(xv, ck[i], dk[i]));

        v16h A0, A1;
        #pragma unroll
        for (int i = 0; i < 16; ++i) {
            A0[i] = (_Float16)h0v[i];
            A1[i] = (_Float16)h0v[16 + i];
        }

        // h0 global store: each VGPR slot-pair is a consecutive-k float2
        #pragma unroll
        for (int f = 0; f < 2; ++f)
          #pragma unroll
          for (int v = 0; v < 8; ++v) {
              const int k = 32*f + 2*v + (v >= 4 ? 8 : 0) + 8*half;
              float2 p = make_float2(h0v[f*16 + 2*v], h0v[f*16 + 2*v + 1]);
              *(float2*)(h0_out + (size_t)r*64 + k) = p;
          }

        // ---- layer 1: 16x64 = (16x64) @ (64x64) via 8 WMMAs --------------
        v8f acc[4];
        #pragma unroll
        for (int t = 0; t < 4; ++t) {
            v8f c = {};
            c = __builtin_amdgcn_wmma_f32_16x16x32_f16(
                    false, A0, false, Bf[t][0], (short)0, c, false, false);
            c = __builtin_amdgcn_wmma_f32_16x16x32_f16(
                    false, A1, false, Bf[t][1], (short)0, c, false, false);
            acc[t] = c;
        }

        // ---- tanh + h1 store + fc partial sums ---------------------------
        float p0[8], p1[8];
        #pragma unroll
        for (int g = 0; g < 8; ++g) { p0[g] = 0.f; p1[g] = 0.f; }

        #pragma unroll
        for (int t = 0; t < 4; ++t) {
            const int col = 16*t + m16;
            #pragma unroll
            for (int g = 0; g < 8; ++g) {
                const float hv  = fast_tanh(acc[t][g] + bs1[t]);
                const int   row = rowbase + g + 8*half;   // C layout: M=g+8*half
                h1_out[(size_t)row*64 + col] = hv;
                p0[g] = fmaf(hv, fw0[t], p0[g]);
                p1[g] = fmaf(hv, fw1[t], p1[g]);
            }
        }

        // ---- reduce fc partials across the 16-lane group, store out ------
        #pragma unroll
        for (int g = 0; g < 8; ++g) {
            float a = p0[g], b = p1[g];
            #pragma unroll
            for (int m = 1; m < 16; m <<= 1) {
                a += __shfl_xor(a, m, 32);
                b += __shfl_xor(b, m, 32);
            }
            if (m16 == 0) {
                const int row = rowbase + g + 8*half;
                *(float2*)(out + (size_t)row*2) = make_float2(a + fb0, b + fb1);
            }
        }
    }
}

extern "C" void kernel_launch(void* const* d_in, const int* in_sizes, int n_in,
                              void* d_out, int out_size, void* d_ws, size_t ws_size,
                              hipStream_t stream) {
    // setup_inputs order: x, W_ih0, b_ih0, W_hh0, b_hh0, W_ih1, b_ih1,
    //                     W_hh1, b_hh1, fc_w, fc_b   (W_hh* unused)
    const float* x     = (const float*)d_in[0];
    const float* W_ih0 = (const float*)d_in[1];
    const float* b_ih0 = (const float*)d_in[2];
    const float* b_hh0 = (const float*)d_in[4];
    const float* W_ih1 = (const float*)d_in[5];
    const float* b_ih1 = (const float*)d_in[6];
    const float* b_hh1 = (const float*)d_in[8];
    const float* fc_w  = (const float*)d_in[9];
    const float* fc_b  = (const float*)d_in[10];

    float* out = (float*)d_out;                    // SEQ*2
    float* h0o = out + (size_t)SEQ_N * 2;          // hidden[0]: SEQ*64
    float* h1o = h0o + (size_t)SEQ_N * 64;         // hidden[1]: SEQ*64

    dim3 grid(1024), block(256);                   // 8192 waves over 62500 tiles
    hipLaunchKernelGGL(rnn_fused_wmma, grid, block, 0, stream,
                       x, W_ih0, b_ih0, b_hh0, W_ih1, b_ih1, b_hh1,
                       fc_w, fc_b, out, h0o, h1o);
}
